// MultiHeadSelfAttention_85985245266523
// MI455X (gfx1250) — matrix operand order repaired and checked by reading, not compiled, not run on hardware
//
#include <hip/hip_runtime.h>
#include <hip/hip_bf16.h>

#ifndef __has_builtin
#define __has_builtin(x) 0
#endif
#if __has_builtin(__builtin_amdgcn_tensor_load_to_lds)
#define USE_TDM 1
#else
#define USE_TDM 0
#endif

// ---------------------------------------------------------------------------
// Types for WMMA fragments (CDNA5 / gfx1250, wave32)
// ---------------------------------------------------------------------------
typedef _Float16     v16h __attribute__((ext_vector_type(16)));
typedef _Float16     v8h  __attribute__((ext_vector_type(8)));
typedef float        v8f  __attribute__((ext_vector_type(8)));
typedef unsigned int v4u  __attribute__((ext_vector_type(4)));
typedef int          v8i  __attribute__((ext_vector_type(8)));
typedef int          v4i  __attribute__((ext_vector_type(4)));

__device__ __forceinline__ v8f wmma16x16x32(v16h a, v16h b, v8f c) {
  // D = A(16x32 f16) * B(32x16 f16) + C(16x16 f32)
  return __builtin_amdgcn_wmma_f32_16x16x32_f16(
      /*neg_a=*/false, a, /*neg_b=*/false, b,
      /*c_mod=*/(short)0, c, /*reuse_a=*/false, /*reuse_b=*/false);
}

union frag_u { struct { v8h lo, hi; } p; v16h v; };

// A fragment 16x32 f16. ISA 7.12.2: lane holds row M=lane&15; per lane the
// 16 halves live at contiguous runs [hi*8 .. hi*8+7] and [16+hi*8 .. +7]
// (hi = lane>=16), so two 16-byte loads suffice. ld must keep rows 16B-aligned.
__device__ __forceinline__ v16h load_a16(const _Float16* p0, int ld, int lane) {
  const int m  = lane & 15;
  const int hi = (lane >> 4) & 1;
  const _Float16* p = p0 + (size_t)m * ld + (hi << 3);
  frag_u u;
  u.p.lo = *(const v8h*)p;
  u.p.hi = *(const v8h*)(p + 16);
  return u.v;
}

// B fragment 32x16 f16 from N-major storage (column n contiguous in K).
// Lane holds column N=lane&15; halves at K runs [hi*8 .. hi*8+7] and
// [16+hi*8 .. +7] (hi = lane>=16), matching A -> two 16-byte loads.
__device__ __forceinline__ v16h load_b16(const _Float16* p0, int ld, int lane) {
  const int n   = lane & 15;
  const int kb0 = ((lane >> 4) & 1) << 3;
  const _Float16* p = p0 + (size_t)n * ld + kb0;
  frag_u u;
  u.p.lo = *(const v8h*)p;
  u.p.hi = *(const v8h*)(p + 16);
  return u.v;
}

// Cross-lane reductions over the 16-lane column group of the WMMA C layout,
// via ds_swizzle_b32 (group-of-32, xor masks 1/2/4/8 stay inside each half).
#define SWZ_F32(v, imm) \
  __int_as_float(__builtin_amdgcn_ds_swizzle(__float_as_int(v), (imm)))

__device__ __forceinline__ float red_max16(float v) {
  v = fmaxf(v, SWZ_F32(v, 0x041F));
  v = fmaxf(v, SWZ_F32(v, 0x081F));
  v = fmaxf(v, SWZ_F32(v, 0x101F));
  v = fmaxf(v, SWZ_F32(v, 0x201F));
  return v;
}
__device__ __forceinline__ float red_sum16(float v) {
  v += SWZ_F32(v, 0x041F);
  v += SWZ_F32(v, 0x081F);
  v += SWZ_F32(v, 0x101F);
  v += SWZ_F32(v, 0x201F);
  return v;
}

// ---------------------------------------------------------------------------
// Problem constants
// ---------------------------------------------------------------------------
constexpr int D_MODEL = 768;
constexpr int N_HEADS = 12;
constexpr int HEAD_DIM = 64;
constexpr int BATCH = 2;
constexpr int TSEQ = 4096;
constexpr int MROWS = BATCH * TSEQ;  // 8192
constexpr size_t QKV_PLANE = (size_t)BATCH * N_HEADS * TSEQ * HEAD_DIM;  // 6291456

#if USE_TDM
// 1-D TDM copy of `elems` f16 from global to LDS, with LDS padding of
// 4 DWORDs after every 32 DWORDs (64 halfs -> row stride 72 halfs).
// D# layout per CDNA5 ISA 8.3/8.4. Issued by one wave; EXEC ignored.
// amdgpu-toolchain (clang-23) 6-arg builtin form.
__device__ __forceinline__ void tdm_load_f16_pad72(unsigned lds_off,
                                                   const _Float16* src,
                                                   unsigned elems) {
  const unsigned long long ga = (unsigned long long)(const void*)src;
  v4u g0;
  g0[0] = 1u;                                    // count=1, user descriptor
  g0[1] = lds_off;                               // lds_addr (bytes)
  g0[2] = (unsigned)(ga & 0xffffffffu);          // global_addr[31:0]
  g0[3] = (unsigned)((ga >> 32) & 0x1ffffffu)    // global_addr[56:32]
          | (2u << 30);                          // type=2 ("image")
  v8i g1;
  // data_size=1 (2B) | pad_enable | pad_interval=4 (32 DW) | pad_amount=3 (4 DW)
  g1[0] = (int)0x07110000u;
  g1[1] = (int)((elems & 0xffffu) << 16);        // tensor_dim0[15:0]
  g1[2] = (int)(((elems >> 16) & 0xffffu) | (1u << 16));  // dim0 hi, tensor_dim1=1
  g1[3] = (int)((elems & 0xffffu) << 16);        // tile_dim0 = elems
  g1[4] = 1;                                     // tile_dim1 = 1
  g1[5] = (int)elems;                            // tensor_dim0_stride
  g1[6] = 0;
  g1[7] = 0;
  const v4i z4 = {0, 0, 0, 0};
  const v8i z8 = {0, 0, 0, 0, 0, 0, 0, 0};
  __builtin_amdgcn_tensor_load_to_lds(g0, g1, z4, z4, z8, 0);
}
#endif

// ---------------------------------------------------------------------------
// Tiled WMMA GEMM:  C[M,N] = A[M,K] * W[N,K]^T
//   MODE 0: A = x (f32), W = w_qkv; write f16 scattered into [s][b][h][t][hd]
//   MODE 1: A = attn-out (f32), W = w_proj; write f32 out + bias
// Block: 256 threads = 8 waves in a 4(m) x 2(n) grid; wave tile 32x64.
// ---------------------------------------------------------------------------
constexpr int BM = 128, BN = 128, BK = 32;
constexpr int LDT = BK + 8;  // 40 halfs = 80 B rows: 16B-aligned, conflict-free

template <int MODE>
__global__ __launch_bounds__(256, 2) void gemm_wmma_kernel(
    const float* __restrict__ A, const float* __restrict__ W,
    const float* __restrict__ bias, _Float16* __restrict__ qkv_out,
    float* __restrict__ out, int M, int N, int K) {
  __shared__ __align__(16) _Float16 sA[BM * LDT];
  __shared__ __align__(16) _Float16 sB[BN * LDT];

  const int tid  = threadIdx.x;
  const int lane = tid & 31;
  const int wave = tid >> 5;
  const int wm   = wave >> 1;  // 0..3
  const int wn   = wave & 1;   // 0..1
  const int rowBlk = blockIdx.y * BM;
  const int colBlk = blockIdx.x * BN;

  v8f acc[2][4] = {};

  // Cooperative tile loaders: 128 rows, 2 threads/row, 16 floats each.
  const int lrow = tid >> 1;
  const int lcol = (tid & 1) << 4;
  const float* pA = A + (size_t)(rowBlk + lrow) * K + lcol;
  const float* pW = W + (size_t)(colBlk + lrow) * K + lcol;
  _Float16* dA = sA + lrow * LDT + lcol;
  _Float16* dB = sB + lrow * LDT + lcol;

  const int kSteps = K / BK;
  for (int ks = 0; ks < kSteps; ++ks) {
#pragma unroll
    for (int j = 0; j < 2; ++j) {
      const float4 fa0 = *(const float4*)(pA + 8 * j);
      const float4 fa1 = *(const float4*)(pA + 8 * j + 4);
      const float4 fw0 = *(const float4*)(pW + 8 * j);
      const float4 fw1 = *(const float4*)(pW + 8 * j + 4);
      const v8h ha = {(_Float16)fa0.x, (_Float16)fa0.y, (_Float16)fa0.z, (_Float16)fa0.w,
                      (_Float16)fa1.x, (_Float16)fa1.y, (_Float16)fa1.z, (_Float16)fa1.w};
      const v8h hw = {(_Float16)fw0.x, (_Float16)fw0.y, (_Float16)fw0.z, (_Float16)fw0.w,
                      (_Float16)fw1.x, (_Float16)fw1.y, (_Float16)fw1.z, (_Float16)fw1.w};
      *(v8h*)(dA + 8 * j) = ha;
      *(v8h*)(dB + 8 * j) = hw;
    }
    pA += BK;
    pW += BK;
    __builtin_prefetch(pA, 0, 1);  // global_prefetch_b8 of next tiles
    __builtin_prefetch(pW, 0, 1);
    __syncthreads();

    v16h af[2], bf[4];
#pragma unroll
    for (int i = 0; i < 2; ++i)
      af[i] = load_a16(sA + (wm * 32 + i * 16) * LDT, LDT, lane);
#pragma unroll
    for (int j = 0; j < 4; ++j)
      bf[j] = load_b16(sB + (wn * 64 + j * 16) * LDT, LDT, lane);
#pragma unroll
    for (int i = 0; i < 2; ++i)
#pragma unroll
      for (int j = 0; j < 4; ++j)
        acc[i][j] = wmma16x16x32(af[i], bf[j], acc[i][j]);
    __syncthreads();
  }

  // Epilogue. C layout: VGPR r -> row r + 8*(lane>=16), col lane&15.
  const int hi = lane >> 4;
  const int n  = lane & 15;
#pragma unroll
  for (int i = 0; i < 2; ++i) {
    const int rbase = rowBlk + wm * 32 + i * 16 + (hi << 3);
#pragma unroll
    for (int j = 0; j < 4; ++j) {
      const int col = colBlk + wn * 64 + j * 16 + n;
      if constexpr (MODE == 0) {
        const int s   = col / D_MODEL;  // 0=Q 1=K 2=V
        const int rem = col - s * D_MODEL;
        const int h   = rem >> 6;
        const int hd  = rem & 63;
#pragma unroll
        for (int r = 0; r < 8; ++r) {
          const int t  = rbase + r;
          const int bb = t >> 12;  // /4096
          const int tt = t & 4095;
          qkv_out[(((size_t)s * BATCH + bb) * N_HEADS + h) * ((size_t)TSEQ * HEAD_DIM) +
                  (size_t)tt * HEAD_DIM + hd] = (_Float16)acc[i][j][r];
        }
      } else {
        const float bv = bias[col];
#pragma unroll
        for (int r = 0; r < 8; ++r)
          out[(size_t)(rbase + r) * N + col] = acc[i][j][r] + bv;
      }
    }
  }
}

// ---------------------------------------------------------------------------
// Flash-style attention. Grid: (T/64, B*H). Block: 128 threads = 4 waves.
// Each wave owns a 16-row Q tile; block shares a 32x64 K tile (kv-major,
// TDM-loaded with LDS padding when available) and a transposed 64x32 V tile.
// ---------------------------------------------------------------------------
constexpr int BKV = 32;
constexpr int LKV = HEAD_DIM + 8;  // 72 halfs = 144 B rows (16B-aligned)
constexpr int LVT = BKV + 8;       // 40 halfs = 80 B rows (16B-aligned)
constexpr int LP  = 40;

__global__ __launch_bounds__(128, 2) void attn_fwd_kernel(
    const _Float16* __restrict__ qkv, const unsigned char* __restrict__ mask,
    float* __restrict__ Ows) {
  __shared__ __align__(16) _Float16 sK[BKV * LKV];       // [kv][d]
  __shared__ __align__(16) _Float16 sVt[HEAD_DIM * LVT]; // [d][kv] (transposed)
  __shared__ __align__(16) _Float16 sP[4 * 16 * LP];     // per-wave P scratch

  const int tid  = threadIdx.x;
  const int lane = tid & 31;
  const int wave = tid >> 5;
  const int bh = blockIdx.y;
  const int b  = bh / N_HEADS;
  const int h  = bh - b * N_HEADS;
  const int q0 = blockIdx.x * 64 + wave * 16;

  const size_t headbase = (size_t)bh * TSEQ * HEAD_DIM;
  const _Float16* Q  = qkv + headbase;
  const _Float16* Kp = qkv + QKV_PLANE + headbase;
  const _Float16* Vp = qkv + 2 * QKV_PLANE + headbase;
  const unsigned char* mk = mask + (size_t)b * TSEQ;

  const int hi = lane >> 4;
  const int n  = lane & 15;

  // Q fragments (A layout), two K=32 chunks: 4 x global 16B loads total.
  v16h qa[2];
  {
    const _Float16* qrow = Q + (size_t)(q0 + n) * HEAD_DIM + (hi << 3);
#pragma unroll
    for (int c = 0; c < 2; ++c) {
      frag_u u;
      u.p.lo = *(const v8h*)(qrow + c * 32);
      u.p.hi = *(const v8h*)(qrow + c * 32 + 16);
      qa[c] = u.v;
    }
  }

  v8f o[4] = {};
  float mi[8], li[8];
#pragma unroll
  for (int r = 0; r < 8; ++r) { mi[r] = -1e30f; li[r] = 0.0f; }

  // Cooperative tile loaders: 32 rows x 64 halfs, 4 threads/row, 16 halfs each.
  const int lrow = tid >> 2;
  const int lcol = (tid & 3) << 4;

#if USE_TDM
  const unsigned sK_off = (unsigned)(uintptr_t)(const void*)sK;
#endif

  for (int kv = 0; kv < TSEQ; kv += BKV) {
#if USE_TDM
    // Tensor Data Mover: 32x64 f16 K tile is row-contiguous in global memory;
    // TDM pads 4 DWORDs per 32 DWORDs -> LDS row stride LKV=72 halfs.
    if (wave == 0) {
      tdm_load_f16_pad72(sK_off, Kp + (size_t)kv * HEAD_DIM, BKV * HEAD_DIM);
      __builtin_amdgcn_s_wait_tensorcnt(0);
    }
#else
    {
      const _Float16* gK = Kp + (size_t)(kv + lrow) * HEAD_DIM + lcol;
      *(v8h*)(sK + lrow * LKV + lcol)     = *(const v8h*)gK;
      *(v8h*)(sK + lrow * LKV + lcol + 8) = *(const v8h*)(gK + 8);
    }
#endif
    // V tile: load coalesced, store transposed [d][kv].
    {
      const _Float16* gV = Vp + (size_t)(kv + lrow) * HEAD_DIM + lcol;
      const v8h va = *(const v8h*)gV;
      const v8h vb = *(const v8h*)(gV + 8);
#pragma unroll
      for (int j = 0; j < 8; ++j) {
        sVt[(lcol + j) * LVT + lrow]     = va[j];
        sVt[(lcol + 8 + j) * LVT + lrow] = vb[j];
      }
    }
    const bool msk0 = mk[kv + n] != 0;
    const bool msk1 = mk[kv + 16 + n] != 0;
    __syncthreads();

    // S = Q * K^T : two 16x16 column tiles, K-dim 64 = 2 chained WMMAs each.
    v8f s0 = {}, s1 = {};
    {
      const v16h b00 = load_b16(sK + 0 * LKV + 0,  LKV, lane);
      const v16h b01 = load_b16(sK + 0 * LKV + 32, LKV, lane);
      const v16h b10 = load_b16(sK + 16 * LKV + 0,  LKV, lane);
      const v16h b11 = load_b16(sK + 16 * LKV + 32, LKV, lane);
      s0 = wmma16x16x32(qa[0], b00, s0);
      s0 = wmma16x16x32(qa[1], b01, s0);
      s1 = wmma16x16x32(qa[0], b10, s1);
      s1 = wmma16x16x32(qa[1], b11, s1);
    }

    // Online softmax; lane owns one column of rows r+8*hi.
    _Float16* pw = sP + wave * 16 * LP;
#pragma unroll
    for (int r = 0; r < 8; ++r) {
      const float a0 = msk0 ? -1e30f : s0[r] * 0.125f;  // 1/sqrt(64)
      const float a1 = msk1 ? -1e30f : s1[r] * 0.125f;
      const float rmx   = red_max16(fmaxf(a0, a1));
      const float mnew  = fmaxf(mi[r], rmx);
      const float alpha = __expf(mi[r] - mnew);
      const float p0 = __expf(a0 - mnew);
      const float p1 = __expf(a1 - mnew);
      li[r] = li[r] * alpha + red_sum16(p0 + p1);
      mi[r] = mnew;
      o[0][r] *= alpha; o[1][r] *= alpha; o[2][r] *= alpha; o[3][r] *= alpha;
      const int prow = r + (hi << 3);
      pw[prow * LP + n]      = (_Float16)p0;  // C-layout -> LDS
      pw[prow * LP + 16 + n] = (_Float16)p1;
    }

    // Re-load P in A layout (same-wave DS ordering) and accumulate O += P*V.
    const v16h pa = load_a16(pw, LP, lane);
#pragma unroll
    for (int t = 0; t < 4; ++t)
      o[t] = wmma16x16x32(pa, load_b16(sVt + (t * 16) * LVT, LVT, lane), o[t]);
    __syncthreads();
  }

  // Normalize and store f32 (B,T,D) for the output projection GEMM.
#pragma unroll
  for (int r = 0; r < 8; ++r) {
    const float inv = 1.0f / li[r];
    const int q = q0 + r + (hi << 3);
    float* dst = Ows + ((size_t)b * TSEQ + q) * D_MODEL + h * HEAD_DIM;
#pragma unroll
    for (int t = 0; t < 4; ++t)
      dst[t * 16 + n] = o[t][r] * inv;
  }
}

// ---------------------------------------------------------------------------
// Launch: QKV GEMM -> attention -> output projection GEMM
// ---------------------------------------------------------------------------
extern "C" void kernel_launch(void* const* d_in, const int* in_sizes, int n_in,
                              void* d_out, int out_size, void* d_ws, size_t ws_size,
                              hipStream_t stream) {
  const float*         x      = (const float*)d_in[0];
  const unsigned char* mask   = (const unsigned char*)d_in[1];
  const float*         w_qkv  = (const float*)d_in[2];
  const float*         w_proj = (const float*)d_in[3];
  const float*         b_proj = (const float*)d_in[4];
  float*               out    = (float*)d_out;

  _Float16* qkv_ws = (_Float16*)d_ws;                         // 3 f16 planes
  const size_t qkv_bytes = 3 * QKV_PLANE * sizeof(_Float16);  // 37,748,736 B
  float* o_ws = (float*)((char*)d_ws + qkv_bytes);            // 25,165,824 B

  // 1) QKV projection: (8192 x 768) * (2304 x 768)^T -> f16 [s][b][h][t][hd]
  gemm_wmma_kernel<0><<<dim3((3 * D_MODEL) / BN, MROWS / BM), 256, 0, stream>>>(
      x, w_qkv, nullptr, qkv_ws, nullptr, MROWS, 3 * D_MODEL, D_MODEL);

  // 2) Attention per head, flash-style.
  attn_fwd_kernel<<<dim3(TSEQ / 64, BATCH * N_HEADS), 128, 0, stream>>>(
      qkv_ws, mask, o_ws);

  // 3) Output projection: (8192 x 768) * (768 x 768)^T + bias -> f32 out
  gemm_wmma_kernel<1><<<dim3(D_MODEL / BN, MROWS / BM), 256, 0, stream>>>(
      o_ws, w_proj, b_proj, nullptr, out, MROWS, D_MODEL, D_MODEL);
}
